// LSTM_23802708755079
// MI455X (gfx1250) — matrix operand order repaired and checked by reading, not compiled, not run on hardware
//
#include <hip/hip_runtime.h>

// ---------------- problem constants ----------------
#define BATCH   4096
#define TSTEPS  100
#define FIN     13
#define HID     100
#define NGATE   400          // 4*H
#define NT      25           // N tiles of 16 covering 400 gate outputs
#define MROWS   16           // batch rows per block (one WMMA M tile)
#define WAVES   8
#define BLOCK   256          // 8 wave32s

// ---------------- packed-weight workspace layout ----------------
// One B-fragment = 32x16 (KxN) fp16 tile, stored lane-contiguous:
// frag*512 + lane*16 halves  (1 KB per fragment, coalesced 32B/lane loads)
#define FRAG_HALVES     512
#define FRAG_WIH0_BASE  0        // 25 frags  (K tile 0: F=13 padded to 32)
#define FRAG_WHH0_BASE  25       // 25*4 frags (H=100 padded to K=128)
#define FRAG_WIH1_BASE  125      // 25*4 frags
#define FRAG_WHH1_BASE  225      // 25*4 frags
#define NFRAGS          325
#define BIAS_OFF_BYTES  (NFRAGS * FRAG_HALVES * 2)   // 332800 B, 16B aligned

typedef __attribute__((ext_vector_type(16))) _Float16 v16h;
typedef __attribute__((ext_vector_type(8)))  float    v8f;

union F16Frag { v16h v; uint4 q[2]; };

// ---------------- weight packing prepass ----------------
// B[k][n] = W[ntile*16+n][ktile*32+k] (fp32 -> fp16), zero padded past Kin.
// Lane layout (mirror of 16-bit A layout): lanes 0-15 : N=lane, halves 0..7 ->
// K 0..7, halves 8..15 -> K 16..23 ; lanes 16-31 : N=lane-16, K 8..15 / 24..31.
__global__ void lstm_pack_weights(const float* __restrict__ Wih0,
                                  const float* __restrict__ Whh0,
                                  const float* __restrict__ Wih1,
                                  const float* __restrict__ Whh1,
                                  _Float16* __restrict__ frags) {
  int gid = blockIdx.x * blockDim.x + threadIdx.x;
  if (gid >= NFRAGS * 32) return;
  int frag = gid >> 5;
  int lane = gid & 31;

  const float* W; int Kin, nt, kt;
  if (frag < 25)        { W = Wih0; Kin = FIN; nt = frag;            kt = 0; }
  else if (frag < 125)  { int f = frag - 25;  W = Whh0; Kin = HID; nt = f >> 2; kt = f & 3; }
  else if (frag < 225)  { int f = frag - 125; W = Wih1; Kin = HID; nt = f >> 2; kt = f & 3; }
  else                  { int f = frag - 225; W = Whh1; Kin = HID; nt = f >> 2; kt = f & 3; }

  int n    = nt * 16 + (lane & 15);
  int koff = (lane < 16) ? 0 : 8;
  _Float16* dst = frags + (size_t)frag * FRAG_HALVES + lane * 16;
#pragma unroll
  for (int j = 0; j < 16; ++j) {
    int klocal = (j < 8) ? (koff + j) : (koff + 16 + (j - 8));
    int k = kt * 32 + klocal;
    float v = (k < Kin) ? W[n * Kin + k] : 0.0f;
    dst[j] = (_Float16)v;
  }
}

__global__ void lstm_pack_bias(const float* __restrict__ bih0, const float* __restrict__ bhh0,
                               const float* __restrict__ bih1, const float* __restrict__ bhh1,
                               float* __restrict__ bsum) {
  int i = blockIdx.x * blockDim.x + threadIdx.x;
  if (i < NGATE)            bsum[i] = bih0[i] + bhh0[i];
  else if (i < 2 * NGATE)   { int j = i - NGATE; bsum[i] = bih1[j] + bhh1[j]; }
}

// ---------------- fragment loaders ----------------
__device__ __forceinline__ v16h load_a_frag(const _Float16* buf, int stride,
                                            int kbase, int lane) {
  // 16-bit A layout: lanes 0-15 hold row M=lane, K {0..7,16..23};
  // lanes 16-31 hold row M=lane-16, K {8..15,24..31}.
  int m  = lane & 15;
  int ks = kbase + ((lane < 16) ? 0 : 8);
  const _Float16* p = buf + m * stride + ks;   // 16B-aligned by construction
  F16Frag f;
  f.q[0] = *(const uint4*)(p);
  f.q[1] = *(const uint4*)(p + 16);
  return f.v;
}

__device__ __forceinline__ v16h load_b_frag(const _Float16* __restrict__ frags,
                                            int frag, int lane) {
  const _Float16* p = frags + (size_t)frag * FRAG_HALVES + lane * 16;
  F16Frag f;
  f.q[0] = *(const uint4*)(p);
  f.q[1] = *(const uint4*)(p + 8);
  return f.v;
}

__device__ __forceinline__ v8f wmma_f16(v16h a, v16h b, v8f c) {
  // (neg_a, A, neg_b, B, c_mod, C, reuse_a, reuse_b)
  return __builtin_amdgcn_wmma_f32_16x16x32_f16(false, a, false, b,
                                                (short)0, c, false, false);
}

// ---------------- fast activations: v_exp_f32 + v_rcp_f32 ----------------
#define LOG2E 1.4426950408889634f
__device__ __forceinline__ float sigm(float v) {
  // sigma(v) = 1 / (1 + 2^(-v*log2e)) : one v_exp_f32 + one v_rcp_f32
  return __builtin_amdgcn_rcpf(1.0f + __builtin_amdgcn_exp2f(-v * LOG2E));
}
__device__ __forceinline__ float fast_tanh(float v) {
  // tanh(v) = 2*sigma(2v) - 1
  return __builtin_amdgcn_fmed3f(2.0f * sigm(2.0f * v) - 1.0f, -1.0f, 1.0f);
}

// ---------------- persistent LSTM kernel ----------------
// One block owns MROWS=16 batch rows for all 100 timesteps.
// h/c state lives in LDS; weights stream from the L2-resident packed buffer.
#define HSTRIDE 136   // h row stride in halves: 272B, 16B aligned, bank-spread
#define XSTRIDE 40    // x row stride in halves: 80B,  16B aligned, bank-spread
#define ZSTRIDE 404   // z row stride in floats: breaks half-wave bank aliasing

__launch_bounds__(BLOCK, 1)
__global__ void lstm_persistent(const float* __restrict__ x,
                                const int*   __restrict__ lengths,
                                const _Float16* __restrict__ frags,
                                const float* __restrict__ bsum,
                                const float* __restrict__ Wfc,
                                const float* __restrict__ bfc,
                                float* __restrict__ out) {
  __shared__ _Float16 h0[MROWS][HSTRIDE];
  __shared__ _Float16 h1[MROWS][HSTRIDE];
  __shared__ _Float16 xb[MROWS][XSTRIDE];
  __shared__ float    zb[MROWS][ZSTRIDE];
  __shared__ float    c0[MROWS][HID];
  __shared__ float    c1[MROWS][HID];
  __shared__ float    bias[2 * NGATE];

  const int tid  = threadIdx.x;
  const int lane = tid & 31;
  const int b0   = blockIdx.x * MROWS;
  // Wave index is uniform across the wave; force it into an SGPR so the tile
  // loops below become scalar-controlled (WMMA requires EXEC all-ones).
  const int wave = __builtin_amdgcn_readfirstlane(tid >> 5);

  // ---- gate-stage geometry (computed ONCE, loop-invariant) ----
  // 16 threads per row; 7 column slots of stride 16 cover HID=100.
  const int gr = tid >> 4;        // row 0..15
  const int gu = tid & 15;        // base unit within the row
  const int myLen = lengths[b0 + gr];   // per-thread length, kept in a register

  // ---- x-staging geometry (computed ONCE) ----
  const int  xact = (tid < MROWS * FIN);
  const int  xr   = xact ? (tid / FIN) : 0;
  const int  xj   = xact ? (tid - xr * FIN) : 0;
  const float* xsrc = x + (size_t)(b0 + xr) * TSTEPS * FIN + xj;

  // ---- init: zero state & padding, stage biases ----
  for (int i = tid; i < MROWS * HSTRIDE; i += BLOCK) {
    (&h0[0][0])[i] = (_Float16)0.0f;
    (&h1[0][0])[i] = (_Float16)0.0f;
  }
  for (int i = tid; i < MROWS * XSTRIDE; i += BLOCK) (&xb[0][0])[i] = (_Float16)0.0f;
  for (int i = tid; i < MROWS * HID; i += BLOCK) {
    (&c0[0][0])[i] = 0.0f;
    (&c1[0][0])[i] = 0.0f;
  }
  for (int i = tid; i < 2 * NGATE; i += BLOCK) bias[i] = bsum[i];
  __syncthreads();

  for (int t = 0; t < TSTEPS; ++t) {
    // ---- stage x_t (fp32 -> fp16, K padded 13->32 with persistent zeros) ----
    if (xact) xb[xr][xj] = (_Float16)(*xsrc);
    xsrc += FIN;
    __syncthreads();

    // ---- layer 0 GEMM: z0 = [x_t | h0] x [Wih0 | Whh0]^T ----
    {
      v16h ax = load_a_frag(&xb[0][0], XSTRIDE, 0, lane);
      v16h ah[4];
#pragma unroll
      for (int kt = 0; kt < 4; ++kt)
        ah[kt] = load_a_frag(&h0[0][0], HSTRIDE, kt * 32, lane);

      for (int nt = wave; nt < NT; nt += WAVES) {
        v8f acc = {};
        acc = wmma_f16(ax, load_b_frag(frags, FRAG_WIH0_BASE + nt, lane), acc);
#pragma unroll
        for (int kt = 0; kt < 4; ++kt)
          acc = wmma_f16(ah[kt],
                         load_b_frag(frags, FRAG_WHH0_BASE + nt * 4 + kt, lane), acc);
        int col   = nt * 16 + (lane & 15);
        int rbase = (lane < 16) ? 0 : 8;
#pragma unroll
        for (int r = 0; r < 8; ++r) zb[rbase + r][col] = acc[r];
      }
    }
    __syncthreads();

    // ---- layer 0 gates (i,f,g,o) + masked state update ----
    if (t < myLen) {
#pragma unroll
      for (int j = 0; j < 7; ++j) {
        int u = gu + j * 16;
        if (u < HID) {
          float zi = zb[gr][u]       + bias[u];
          float zf = zb[gr][u + 100] + bias[u + 100];
          float zg = zb[gr][u + 200] + bias[u + 200];
          float zo = zb[gr][u + 300] + bias[u + 300];
          float c  = sigm(zf) * c0[gr][u] + sigm(zi) * fast_tanh(zg);
          c0[gr][u] = c;
          h0[gr][u] = (_Float16)(sigm(zo) * fast_tanh(c));
        }
      }
    }
    __syncthreads();

    // ---- layer 1 GEMM: z1 = [h0 | h1] x [Wih1 | Whh1]^T ----
    {
      v16h ah[8];
#pragma unroll
      for (int kt = 0; kt < 4; ++kt)
        ah[kt] = load_a_frag(&h0[0][0], HSTRIDE, kt * 32, lane);
#pragma unroll
      for (int kt = 0; kt < 4; ++kt)
        ah[4 + kt] = load_a_frag(&h1[0][0], HSTRIDE, kt * 32, lane);

      for (int nt = wave; nt < NT; nt += WAVES) {
        // Two independent accumulation chains (4 deep each) so the matrix
        // pipe can overlap them instead of serializing one 8-deep D->C chain.
        v8f acc0 = {};
        v8f acc1 = {};
#pragma unroll
        for (int kt = 0; kt < 4; ++kt)
          acc0 = wmma_f16(ah[kt],
                          load_b_frag(frags, FRAG_WIH1_BASE + nt * 4 + kt, lane), acc0);
#pragma unroll
        for (int kt = 0; kt < 4; ++kt)
          acc1 = wmma_f16(ah[4 + kt],
                          load_b_frag(frags, FRAG_WHH1_BASE + nt * 4 + kt, lane), acc1);
        v8f acc = acc0 + acc1;
        int col   = nt * 16 + (lane & 15);
        int rbase = (lane < 16) ? 0 : 8;
#pragma unroll
        for (int r = 0; r < 8; ++r) zb[rbase + r][col] = acc[r];
      }
    }
    __syncthreads();

    // ---- layer 1 gates + masked state update ----
    if (t < myLen) {
#pragma unroll
      for (int j = 0; j < 7; ++j) {
        int u = gu + j * 16;
        if (u < HID) {
          float zi = zb[gr][u]       + bias[NGATE + u];
          float zf = zb[gr][u + 100] + bias[NGATE + u + 100];
          float zg = zb[gr][u + 200] + bias[NGATE + u + 200];
          float zo = zb[gr][u + 300] + bias[NGATE + u + 300];
          float c  = sigm(zf) * c1[gr][u] + sigm(zi) * fast_tanh(zg);
          c1[gr][u] = c;
          h1[gr][u] = (_Float16)(sigm(zo) * fast_tanh(c));
        }
      }
    }
    __syncthreads();
  }

  // ---- final projection: out[b] = h1[b,:] . Wfc + bfc ----
  if (tid < MROWS) {
    float s = bfc[0];
#pragma unroll 4
    for (int k = 0; k < HID; ++k) s += (float)h1[tid][k] * Wfc[k];
    out[b0 + tid] = s;
  }
}

// ---------------- launcher ----------------
extern "C" void kernel_launch(void* const* d_in, const int* in_sizes, int n_in,
                              void* d_out, int out_size, void* d_ws, size_t ws_size,
                              hipStream_t stream) {
  const float* x      = (const float*)d_in[0];
  const int*   lens   = (const int*)  d_in[1];
  const float* Wih0   = (const float*)d_in[2];
  const float* Whh0   = (const float*)d_in[3];
  const float* bih0   = (const float*)d_in[4];
  const float* bhh0   = (const float*)d_in[5];
  const float* Wih1   = (const float*)d_in[6];
  const float* Whh1   = (const float*)d_in[7];
  const float* bih1   = (const float*)d_in[8];
  const float* bhh1   = (const float*)d_in[9];
  const float* Wfc    = (const float*)d_in[10];
  const float* bfc    = (const float*)d_in[11];

  _Float16* frags = (_Float16*)d_ws;
  float*    bsum  = (float*)((char*)d_ws + BIAS_OFF_BYTES);

  lstm_pack_weights<<<(NFRAGS * 32 + 255) / 256, 256, 0, stream>>>(
      Wih0, Whh0, Wih1, Whh1, frags);
  lstm_pack_bias<<<(2 * NGATE + 255) / 256, 256, 0, stream>>>(
      bih0, bhh0, bih1, bhh1, bsum);
  lstm_persistent<<<BATCH / MROWS, BLOCK, 0, stream>>>(
      x, lens, frags, bsum, Wfc, bfc, (float*)d_out);
}
